// FBPinn_86285892977067
// MI455X (gfx1250) — matrix operand order repaired and checked by reading, not compiled, not run on hardware
//
#include <hip/hip_runtime.h>
#include <hip/hip_bf16.h>

typedef __attribute__((ext_vector_type(16))) _Float16 v16h;
typedef __attribute__((ext_vector_type(8)))  float    v8f;
typedef __attribute__((ext_vector_type(4)))  float    v4f;

#define NW   16
#define NEUR 32
#define MT   4      // 16-point sub-tiles per wave -> 64 points / wave

__device__ __forceinline__ float fast_tanh(float x) {
#if __has_builtin(__builtin_amdgcn_tanhf)
    return __builtin_amdgcn_tanhf(x);     // v_tanh_f32 (TRANS unit, co-executes with WMMA)
#else
    return tanhf(x);
#endif
}

__device__ __forceinline__ float fast_sig(float t) {
    // sigmoid(t) = 0.5*tanh(0.5t) + 0.5
    return fmaf(0.5f, fast_tanh(0.5f * t), 0.5f);
}

// Pre-pass: convert W2 (NW x NEUR x NEUR fp32, row-major) to f16 in workspace.
// Packed WMMA B layout: half at ((i*2 + n/16)*32 + lane)*16 + s, lane = 16*h + n%16,
// k = table order (h,s).  Lane l loads W2h[(i*64 + l)*16 + {0..15}] (B0) and + 512 (B1).
__global__ void fbpinn_convert_w2(const float* __restrict__ W2,
                                  _Float16* __restrict__ W2h, int n) {
    int idx = blockIdx.x * blockDim.x + threadIdx.x;
    if (idx < n) { int wi = idx >> 10, k = (idx >> 5) & 31, nn = idx & 31; int ln = ((k >> 3) & 1) * 16 + (nn & 15); int s = (k & 7) + ((k >> 4) << 3); W2h[((wi * 2 + (nn >> 4)) * 32 + ln) * 16 + s] = (_Float16)W2[idx]; }
}

__global__ __launch_bounds__(256)
void fbpinn_forward_kernel(const float* __restrict__ x,
                           const float* __restrict__ W1,
                           const float* __restrict__ b1,
                           const _Float16* __restrict__ W2h,
                           const float* __restrict__ b2,
                           const float* __restrict__ W3,
                           const float* __restrict__ b3,
                           float* __restrict__ out, int N)
{
    const int lane = threadIdx.x & 31;
    const int r    = lane & 15;        // point-in-subtile for A rows / neuron id for C cols
    const int half = lane >> 4;        // 0: K 0-7,16-23 ; 1: K 8-15,24-31
    const int c0   = half * 8;

    const long long gw   = (long long)blockIdx.x * (blockDim.x >> 5) + (threadIdx.x >> 5);
    const long long base = gw * (16LL * MT);
    if (base >= N) return;             // wave-uniform: EXEC stays all-ones

    // Load x per sub-tile (clamped so partial tail tiles keep EXEC full)
    float xp[MT];
#pragma unroll
    for (int t = 0; t < MT; ++t) {
        long long idx = base + t * 16 + r;
        if (idx > N - 1) idx = N - 1;
        xp[t] = x[idx];
    }

    float acc[MT][8];
#pragma unroll
    for (int t = 0; t < MT; ++t)
#pragma unroll
        for (int v = 0; v < 8; ++v) acc[t][v] = 0.0f;

    const float w    = 0.0625f;        // domain width / NW
    const float invs = 200.0f;         // 1 / SIGMA

    for (int i = 0; i < NW; ++i) {
        // ---- static subdomain geometry (matches python _geometry) ----
        float sub0 = (i == 0)      ? 0.0f : ((float)i - 0.125f) * w;
        float sub1 = (i == NW - 1) ? 1.0f : ((float)i + 1.125f) * w;
        float mean   = 0.5f * (sub0 + sub1);
        float invstd = 2.0f / (sub1 - sub0);
        float mo0 = (float)i * w;          // MO[i]  == i*w exactly
        float mo1 = (float)(i + 1) * w;    // MO[i+1]

        // ---- per-window weights ----
        const float* W1p = W1 + i * NEUR + c0;
        v4f w1a0 = *(const v4f*)(W1p);          // K = c0..c0+3
        v4f w1a1 = *(const v4f*)(W1p + 4);      // K = c0+4..c0+7
        v4f w1b0 = *(const v4f*)(W1p + 16);     // K = c0+16..c0+19
        v4f w1b1 = *(const v4f*)(W1p + 20);     // K = c0+20..c0+23
        const float* b1p = b1 + i * NEUR + c0;
        v4f b1a0 = *(const v4f*)(b1p);
        v4f b1a1 = *(const v4f*)(b1p + 4);
        v4f b1b0 = *(const v4f*)(b1p + 16);
        v4f b1b1 = *(const v4f*)(b1p + 20);

        const _Float16* Bp = W2h + ((long long)(i * 2 * NEUR + lane)) * 16;
        v16h B0 = *(const v16h*)(Bp);           // W2[k=table(half,s)][n=r]
        v16h B1 = *(const v16h*)(Bp + 512);     // W2[k=table(half,s)][n=r+16]

        float b2a = b2[i * NEUR + r];           // n = r       (C0 column bias)
        float b2b = b2[i * NEUR + r + 16];      // n = r + 16  (C1 column bias)
        float w3a = W3[i * NEUR + r];
        float w3b = W3[i * NEUR + r + 16];
        float b3v = b3[i];

        v8f cin0, cin1;
#pragma unroll
        for (int v = 0; v < 8; ++v) { cin0[v] = b2a; cin1[v] = b2b; }

#pragma unroll
        for (int t = 0; t < MT; ++t) {
            float xv = xp[t];
            float xn = (xv - mean) * invstd;

            // ---- layer 1: this lane's 16 assigned K slots ----
            v4f ha0 = xn * w1a0 + b1a0;
            v4f ha1 = xn * w1a1 + b1a1;
            v4f hb0 = xn * w1b0 + b1b0;
            v4f hb1 = xn * w1b1 + b1b1;
            v16h a;
#pragma unroll
            for (int j = 0; j < 4; ++j) {
                a[j]      = (_Float16)fast_tanh(ha0[j]);
                a[4 + j]  = (_Float16)fast_tanh(ha1[j]);
                a[8 + j]  = (_Float16)fast_tanh(hb0[j]);
                a[12 + j] = (_Float16)fast_tanh(hb1[j]);
            }

            // ---- window weight for this lane's own point (masked) ----
            float win = fast_sig((xv - mo0) * invs) * fast_sig((mo1 - xv) * invs);
            win = (win > 1e-5f) ? win : 0.0f;

            // ---- layer 2 on the matrix pipe: h(16x32) @ W2(32x32) + b2 ----
            v8f cv0 = __builtin_amdgcn_wmma_f32_16x16x32_f16(
                false, a, false, B0, (short)0, cin0, false, false);
            v8f cv1 = __builtin_amdgcn_wmma_f32_16x16x32_f16(
                false, a, false, B1, (short)0, cin1, false, false);

            // ---- tanh + layer 3 partials, weighted by the (shuffled) window ----
#pragma unroll
            for (int v = 0; v < 8; ++v) {
                // C row M = v + 8*half -> that point's window weight lives in lane (v + 8*half)
                float wv = __shfl(win, v + c0, 32);
                float t0 = fast_tanh(cv0[v]);
                float t1 = fast_tanh(cv1[v]);
                float part = fmaf(t0, w3a, t1 * w3b);
                part = (r == 0) ? (part + b3v) : part;   // b3 added exactly once per point
                acc[t][v] += wv * part;
            }
        }
    }

    // ---- reduce over the 16 neurons-per-half (lanes), apply hard constraint, store ----
#pragma unroll
    for (int t = 0; t < MT; ++t) {
#pragma unroll
        for (int v = 0; v < 8; ++v) {
            float s = acc[t][v];
            s += __shfl_xor(s, 1, 32);
            s += __shfl_xor(s, 2, 32);
            s += __shfl_xor(s, 4, 32);
            s += __shfl_xor(s, 8, 32);
            acc[t][v] = s;
        }
        // lane picks acc[r & 7] via a 3-level select tree (no dynamic indexing)
        float a0 = (r & 1) ? acc[t][1] : acc[t][0];
        float a1 = (r & 1) ? acc[t][3] : acc[t][2];
        float a2 = (r & 1) ? acc[t][5] : acc[t][4];
        float a3 = (r & 1) ? acc[t][7] : acc[t][6];
        float s0 = (r & 2) ? a1 : a0;
        float s1 = (r & 2) ? a3 : a2;
        float val = (r & 4) ? s1 : s0;

        val *= fast_tanh(15.0f * xp[t]);     // hard constraint tanh(omega*x)

        long long idx = base + t * 16 + r;
        bool doStore = half ? (r >= 8) : (r < 8);   // lanes 0-7 -> pts 0-7, lanes 24-31 -> pts 8-15
        if (doStore && idx < N) out[idx] = val;
    }
}

extern "C" void kernel_launch(void* const* d_in, const int* in_sizes, int n_in,
                              void* d_out, int out_size, void* d_ws, size_t ws_size,
                              hipStream_t stream) {
    const float* x  = (const float*)d_in[0];
    const float* W1 = (const float*)d_in[1];
    const float* b1 = (const float*)d_in[2];
    const float* W2 = (const float*)d_in[3];
    const float* b2 = (const float*)d_in[4];
    const float* W3 = (const float*)d_in[5];
    const float* b3 = (const float*)d_in[6];
    float* out = (float*)d_out;
    const int N = in_sizes[0];

    _Float16* W2h = (_Float16*)d_ws;       // 16*32*32 halfs = 32 KB
    const int nw2 = NW * NEUR * NEUR;
    fbpinn_convert_w2<<<(nw2 + 255) / 256, 256, 0, stream>>>(W2, W2h, nw2);

    const int pointsPerBlock = 8 * 16 * MT;    // 8 waves * 64 points
    const int grid = (N + pointsPerBlock - 1) / pointsPerBlock;
    fbpinn_forward_kernel<<<grid, 256, 0, stream>>>(x, W1, b1, W2h, b2, W3, b3, out, N);
}
